// MixerBlock_64476049047753
// MI455X (gfx1250) — matrix operand order repaired and checked by reading, not compiled, not run on hardware
//
#include <hip/hip_runtime.h>
#include <hip/hip_bf16.h>

// ---------------------------------------------------------------------------
// MixerBlock for MI455X (gfx1250, wave32, WMMA)
//   B=4 S=2048 D=1024 NH=16 H2=8 F=64 EXP=4
// Pipeline:
//   ln1(x) -> bf16 A
//   GEMM1: proj = A @ Wp' + bp            (v_wmma_f32_16x16x32_bf16)
//   scan : diagonal linear recurrence -> mix (bf16)
//   GEMM2: y = x + mix @ Wo + bo
//   ln2(y) -> bf16
//   GEMM3: t = gelu(ln2 @ W1 + c1) (bf16)
//   GEMM4: out = y + t @ W2 + c2
// Weights pre-transposed to [N][K] bf16 so A and B fragments are contiguous
// ds_load_b128 pairs. Global->LDS staging uses CDNA5 async copies
// (GLOBAL_LOAD_ASYNC_TO_LDS_B128 / ASYNCcnt), double-buffered so the async
// unit streams tile k+1 while WMMA consumes tile k.
// ---------------------------------------------------------------------------

typedef __attribute__((ext_vector_type(16))) __bf16 v16bf;
typedef __attribute__((ext_vector_type(8)))  float  v8f;

#define BQ 4
#define SQ 2048
#define DQ 1024
#define H2Q 8
#define FQ 64
#define TOKENS (BQ * SQ)      // 8192
#define DHID (4 * DQ)         // 4096

__device__ __forceinline__ unsigned short f2bf(float f) {
    unsigned int u = __builtin_bit_cast(unsigned int, f);
    unsigned int r = u + 0x7FFFu + ((u >> 16) & 1u);
    return (unsigned short)(r >> 16);
}

// Per-lane 16-byte async copy global -> LDS (ASYNCcnt). LDS address = low 32
// bits of the generic pointer (flat->LDS aperture truncates to addr[31:0]).
__device__ __forceinline__ void async_copy16(unsigned short* lds_dst,
                                             const unsigned short* gsrc) {
    unsigned lds_off = (unsigned)(uintptr_t)lds_dst;
    asm volatile("global_load_async_to_lds_b128 %0, %1, off"
                 :: "v"(lds_off), "v"(gsrc)
                 : "memory");
}
__device__ __forceinline__ void async_wait0() {
    asm volatile("s_wait_asynccnt 0" ::: "memory");
}
// ASYNCcnt retires in order: <=4 outstanding means everything except the 4
// just-issued next-tile copies has completed.
__device__ __forceinline__ void async_wait4() {
    asm volatile("s_wait_asynccnt 4" ::: "memory");
}

// -------------------------------- LN -> bf16 -------------------------------
__global__ __launch_bounds__(256) void ln_bf16_kernel(
    const float* __restrict__ x, const float* __restrict__ g,
    const float* __restrict__ b, unsigned short* __restrict__ out)
{
    const int t   = blockIdx.x;
    const int tid = threadIdx.x;
    const float* xt = x + (size_t)t * DQ;
    __shared__ float red[256];

    float vals[4];
    float s = 0.f;
#pragma unroll
    for (int i = 0; i < 4; ++i) { vals[i] = xt[tid + i * 256]; s += vals[i]; }
    red[tid] = s; __syncthreads();
    for (int off = 128; off; off >>= 1) {
        if (tid < off) red[tid] += red[tid + off];
        __syncthreads();
    }
    const float mean = red[0] * (1.0f / (float)DQ);
    __syncthreads();

    float s2 = 0.f;
#pragma unroll
    for (int i = 0; i < 4; ++i) { float d = vals[i] - mean; s2 += d * d; }
    red[tid] = s2; __syncthreads();
    for (int off = 128; off; off >>= 1) {
        if (tid < off) red[tid] += red[tid + off];
        __syncthreads();
    }
    const float rstd = rsqrtf(red[0] * (1.0f / (float)DQ) + 1e-5f);

#pragma unroll
    for (int i = 0; i < 4; ++i) {
        const int c = tid + i * 256;
        out[(size_t)t * DQ + c] = f2bf((vals[i] - mean) * rstd * g[c] + b[c]);
    }
}

// --------------------------- weight conversions ----------------------------
// Tiled transpose + f32->bf16: src [K][N] f32 -> dst [N][K] bf16.
__global__ __launch_bounds__(256) void convT_bf16_kernel(
    const float* __restrict__ src, unsigned short* __restrict__ dst,
    int K, int N)
{
    __shared__ float tile[32][33];
    const int k0 = blockIdx.y * 32;
    const int n0 = blockIdx.x * 32;
    const int tx = threadIdx.x & 31;
    const int ty = threadIdx.x >> 5;          // 0..7
#pragma unroll
    for (int i = 0; i < 4; ++i)
        tile[ty + i * 8][tx] = src[(size_t)(k0 + ty + i * 8) * N + n0 + tx];
    __syncthreads();
#pragma unroll
    for (int i = 0; i < 4; ++i)
        dst[(size_t)(n0 + ty + i * 8) * K + k0 + tx] = f2bf(tile[tx][ty + i * 8]);
}

// Wp [NH, D, F] -> WpT [NH*F][D] bf16: WpT[h*F+f][d] = Wp[h, d, f]
__global__ __launch_bounds__(256) void conv_wpT_kernel(
    const float* __restrict__ wp, unsigned short* __restrict__ dst)
{
    int i = blockIdx.x * 256 + threadIdx.x;     // i over D*D, i = col*D + d
    if (i >= DQ * DQ) return;
    int d   = i & (DQ - 1);
    int col = i >> 10;
    int h   = col >> 6;            // /F
    int f   = col & (FQ - 1);
    dst[i] = f2bf(wp[((size_t)h * DQ + d) * FQ + f]);
}

// ------------------------------- WMMA GEMM ---------------------------------
// C[M,N] = epilogue(A[M,K] @ B[K,N] + bias [+ res]) with B given as BT[N][K].
// A, BT bf16 row-major (ushort bits). Block tile 128x128, BK=32, 8 waves,
// double-buffered LDS with async global->LDS copies.
#define GBM 128
#define GBN 128
#define GBK 32
#define LDT_S 40                 // 32 + 8 pad (shorts); row stride 80B = 5*16B
#define TILE_S (GBM * LDT_S)     // shorts per buffer (10240B)

template <bool RES, bool GELU, bool OUTBF16>
__global__ __launch_bounds__(256) void gemm_wmma_bf16(
    const unsigned short* __restrict__ A, const unsigned short* __restrict__ BT,
    const float* __restrict__ bias, const float* __restrict__ res,
    float* __restrict__ Cf, unsigned short* __restrict__ Cb,
    int M, int N, int K)
{
    __shared__ unsigned short sA[2 * TILE_S];
    __shared__ unsigned short sB[2 * TILE_S];

    const int tid  = threadIdx.x;
    const int lane = tid & 31;
    const int wave = tid >> 5;
    const int waveM = wave >> 2;   // 0..1  (64 rows each)
    const int waveN = wave & 3;    // 0..3  (32 cols each)
    const int lm = lane & 15;
    const int kh = lane >> 4;      // K-half owned by this lane

    const int rowBase = blockIdx.y * GBM;
    const int colBase = blockIdx.x * GBN;

    // Straight-line staging indices: each thread moves 2 A + 2 B chunks/tile.
    const int r0  = tid >> 2;          // 0..63
    const int r1  = r0 + 64;           // 64..127
    const int cc0 = (tid & 3) * 8;     // k-offset in shorts (16B chunks)

    const unsigned short* gA0 = A  + (size_t)(rowBase + r0) * K + cc0;
    const unsigned short* gA1 = A  + (size_t)(rowBase + r1) * K + cc0;
    const unsigned short* gB0 = BT + (size_t)(colBase + r0) * K + cc0;
    const unsigned short* gB1 = BT + (size_t)(colBase + r1) * K + cc0;

    auto issue = [&](int buf, int k0) {
        unsigned short* a = sA + buf * TILE_S;
        unsigned short* b = sB + buf * TILE_S;
        async_copy16(&a[r0 * LDT_S + cc0], gA0 + k0);
        async_copy16(&a[r1 * LDT_S + cc0], gA1 + k0);
        async_copy16(&b[r0 * LDT_S + cc0], gB0 + k0);
        async_copy16(&b[r1 * LDT_S + cc0], gB1 + k0);
    };

    v8f acc[4][2] = {};
    const int nk = K / GBK;

    issue(0, 0);
    for (int i = 0; i < nk; ++i) {
        __syncthreads();                       // all reads of buf (i+1)&1 done
        if (i + 1 < nk) {                      // prefetch next tile (async)
            issue((i + 1) & 1, (i + 1) * GBK);
            async_wait4();                     // tile i landed (in-order)
        } else {
            async_wait0();
        }
        __syncthreads();                       // tile i visible to all waves

        const unsigned short* a = sA + (i & 1) * TILE_S;
        const unsigned short* b = sB + (i & 1) * TILE_S;

        // ---- fragments (ISA 7.12.2 layouts); all reads are b128-friendly ----
        v16bf bfrag[2];
#pragma unroll
        for (int nt = 0; nt < 2; ++nt) {
            const int n = waveN * 32 + nt * 16 + lm;
            const int base = n * LDT_S + kh * 8;    // k 8kh..8kh+7, 16+8kh..16+8kh+7
#pragma unroll
            for (int j = 0; j < 16; ++j)
                bfrag[nt][j] = __builtin_bit_cast(__bf16, b[base + ((j < 8) ? j : j + 8)]);
        }
        v16bf afrag[4];
#pragma unroll
        for (int mt = 0; mt < 4; ++mt) {
            const int r = waveM * 64 + mt * 16 + lm;
#pragma unroll
            for (int v = 0; v < 8; ++v) {
                const int kb = ((v < 4) ? 2 * v : 16 + 2 * (v - 4)) + 8 * kh;
                afrag[mt][2 * v]     = __builtin_bit_cast(__bf16, a[r * LDT_S + kb]);
                afrag[mt][2 * v + 1] = __builtin_bit_cast(__bf16, a[r * LDT_S + kb + 1]);
            }
        }
        // ---- 8 WMMAs ----
#pragma unroll
        for (int mt = 0; mt < 4; ++mt)
#pragma unroll
            for (int nt = 0; nt < 2; ++nt)
                acc[mt][nt] = __builtin_amdgcn_wmma_f32_16x16x32_bf16(
                    false, afrag[mt], false, bfrag[nt],
                    (short)0, acc[mt][nt], false, false);
    }

    // ------------------------------ epilogue -------------------------------
#pragma unroll
    for (int mt = 0; mt < 4; ++mt) {
#pragma unroll
        for (int nt = 0; nt < 2; ++nt) {
            const int col = colBase + waveN * 32 + nt * 16 + lm;
            const float bv = bias[col];
#pragma unroll
            for (int r = 0; r < 8; ++r) {
                const int row = rowBase + waveM * 64 + mt * 16 + r + kh * 8;
                float v = acc[mt][nt][r] + bv;
                if (RES) v += res[(size_t)row * N + col];
                if (GELU) {
                    const float x3 = v * v * v;
                    v = 0.5f * v * (1.0f + tanhf(0.7978845608028654f * (v + 0.044715f * x3)));
                }
                if (OUTBF16) Cb[(size_t)row * N + col] = f2bf(v);
                else         Cf[(size_t)row * N + col] = v;
            }
        }
    }
}

// -------------------------- diagonal linear scan ---------------------------
// 4096 independent channels (b, d). d<512 -> col heads, else row heads.
__global__ __launch_bounds__(256) void scan_mix_kernel(
    const float* __restrict__ proj,
    const float* __restrict__ wc, const float* __restrict__ bc,
    const float* __restrict__ wr, const float* __restrict__ br,
    const float* __restrict__ dcol, const float* __restrict__ drow,
    unsigned short* __restrict__ mixb)
{
    const int tid = blockIdx.x * 256 + threadIdx.x;      // 0..4095
    const int b = tid >> 10;
    const int d = tid & (DQ - 1);

    float cache = 0.f;
    if (d < H2Q * FQ) {
        const int h = d >> 6;
        const float dec = powf(fminf(fmaxf(dcol[h], 0.9f), 1.0f), 0.25f);
        const float* w = wc + (size_t)h * SQ;
        const float* bb = bc + (size_t)h * SQ;
        for (int s = 0; s < SQ; ++s) {
            const size_t idx = ((size_t)(b * SQ + s) << 10) + d;
            cache = proj[idx] + dec * cache;
            mixb[idx] = f2bf(w[s] * cache + bb[s]);
        }
    } else {
        const int h = (d - H2Q * FQ) >> 6;
        const float dec = powf(fminf(fmaxf(drow[h], 0.9f), 1.0f), 0.25f);
        const float* w = wr + (size_t)h * SQ;
        const float* bb = br + (size_t)h * SQ;
        for (int s = 0; s < SQ; ++s) {
            const size_t idx = ((size_t)(b * SQ + s) << 10) + d;
            cache = w[s] * proj[idx] + dec * cache;
            mixb[idx] = f2bf(cache + bb[s]);
        }
    }
}

// --------------------------------- launch ----------------------------------
extern "C" void kernel_launch(void* const* d_in, const int* in_sizes, int n_in,
                              void* d_out, int out_size, void* d_ws, size_t ws_size,
                              hipStream_t stream)
{
    const float* x    = (const float*)d_in[0];
    const float* g1   = (const float*)d_in[1];
    const float* b1   = (const float*)d_in[2];
    const float* g2   = (const float*)d_in[3];
    const float* b2   = (const float*)d_in[4];
    const float* Wp   = (const float*)d_in[5];
    const float* bp   = (const float*)d_in[6];
    const float* Wo   = (const float*)d_in[7];
    const float* bo   = (const float*)d_in[8];
    const float* wcol = (const float*)d_in[9];
    const float* bcol = (const float*)d_in[10];
    const float* wrow = (const float*)d_in[11];
    const float* brow = (const float*)d_in[12];
    const float* dcol = (const float*)d_in[13];
    const float* drow = (const float*)d_in[14];
    const float* W1   = (const float*)d_in[15];
    const float* c1   = (const float*)d_in[16];
    const float* W2   = (const float*)d_in[17];
    const float* c2   = (const float*)d_in[18];
    float* out = (float*)d_out;

    // workspace layout (bytes)
    char* ws = (char*)d_ws;
    constexpr size_t SZ_HA   = (size_t)TOKENS * DQ * 2;   // 16 MiB  bf16 ln1
    constexpr size_t SZ_WP   = (size_t)DQ * DQ * 2;       //  2 MiB  WpT
    constexpr size_t SZ_WO   = (size_t)DQ * DQ * 2;       //  2 MiB  WoT
    constexpr size_t SZ_W1   = (size_t)DQ * DHID * 2;     //  8 MiB  W1T
    constexpr size_t SZ_W2   = (size_t)DHID * DQ * 2;     //  8 MiB  W2T
    constexpr size_t SZ_PROJ = (size_t)TOKENS * DQ * 4;   // 32 MiB  f32
    constexpr size_t SZ_MIX  = (size_t)TOKENS * DQ * 2;   // 16 MiB  bf16
    constexpr size_t SZ_Y    = (size_t)TOKENS * DQ * 4;   // 32 MiB  f32
    constexpr size_t SZ_H2   = (size_t)TOKENS * DQ * 2;   // 16 MiB  bf16
    // t (bf16 [8192,4096]) follows

    size_t off = 0;
    unsigned short* hA   = (unsigned short*)(ws + off); off += SZ_HA;
    unsigned short* WpT  = (unsigned short*)(ws + off); off += SZ_WP;
    unsigned short* WoT  = (unsigned short*)(ws + off); off += SZ_WO;
    unsigned short* W1T  = (unsigned short*)(ws + off); off += SZ_W1;
    unsigned short* W2T  = (unsigned short*)(ws + off); off += SZ_W2;
    float*          proj = (float*)         (ws + off); off += SZ_PROJ;
    unsigned short* mixb = (unsigned short*)(ws + off); off += SZ_MIX;
    float*          y    = (float*)         (ws + off); off += SZ_Y;
    unsigned short* h2b  = (unsigned short*)(ws + off); off += SZ_H2;
    unsigned short* tb   = (unsigned short*)(ws + off);

    // 1) LN1 -> bf16
    ln_bf16_kernel<<<TOKENS, 256, 0, stream>>>(x, g1, b1, hA);

    // 2) weight conversions (all transposed to [N][K] bf16)
    conv_wpT_kernel<<<(DQ * DQ + 255) / 256, 256, 0, stream>>>(Wp, WpT);
    convT_bf16_kernel<<<dim3(DQ / 32, DQ / 32), 256, 0, stream>>>(Wo, WoT, DQ, DQ);
    convT_bf16_kernel<<<dim3(DHID / 32, DQ / 32), 256, 0, stream>>>(W1, W1T, DQ, DHID);
    convT_bf16_kernel<<<dim3(DQ / 32, DHID / 32), 256, 0, stream>>>(W2, W2T, DHID, DQ);

    // 3) GEMM1: proj = hA @ Wp' + bp
    gemm_wmma_bf16<false, false, false><<<dim3(DQ / GBN, TOKENS / GBM), 256, 0, stream>>>(
        hA, WpT, bp, nullptr, proj, nullptr, TOKENS, DQ, DQ);

    // 4) recurrence -> mix (bf16)
    scan_mix_kernel<<<(BQ * DQ) / 256, 256, 0, stream>>>(
        proj, wcol, bcol, wrow, brow, dcol, drow, mixb);

    // 5) GEMM2: y = x + mix @ Wo + bo
    gemm_wmma_bf16<true, false, false><<<dim3(DQ / GBN, TOKENS / GBM), 256, 0, stream>>>(
        mixb, WoT, bo, x, y, nullptr, TOKENS, DQ, DQ);

    // 6) LN2 -> bf16
    ln_bf16_kernel<<<TOKENS, 256, 0, stream>>>(y, g2, b2, h2b);

    // 7) GEMM3: t = gelu(h2 @ W1 + c1)  (bf16 out)
    gemm_wmma_bf16<false, true, true><<<dim3(DHID / GBN, TOKENS / GBM), 256, 0, stream>>>(
        h2b, W1T, c1, nullptr, nullptr, tb, TOKENS, DHID, DQ);

    // 8) GEMM4: out = y + t @ W2 + c2
    gemm_wmma_bf16<true, false, false><<<dim3(DQ / GBN, TOKENS / GBM), 256, 0, stream>>>(
        tb, W2T, c2, y, out, nullptr, TOKENS, DQ, DHID);
}
